// CRAMModel_46067819216972
// MI455X (gfx1250) — matrix operand order repaired and checked by reading, not compiled, not run on hardware
//
#include <hip/hip_runtime.h>
#include <hip/hip_bf16.h>

// ---------------------------------------------------------------------------
// CRAM model forward for MI455X (gfx1250, wave32, WMMA).
// Compute-bound (~350 GFLOP vs ~200MB) -> all GEMMs on v_wmma_f32_16x16x32_bf16
// with pre-transposed bf16 weights, double-buffered LDS, b128 fragment loads,
// and CDNA5 async global->LDS staging for bf16 tiles. Retention recurrence
// runs as a 3-pass chunked parallel scan (carry decays as 0.5^t).
// ---------------------------------------------------------------------------

typedef __attribute__((ext_vector_type(16))) __bf16 v16bf;
typedef __attribute__((ext_vector_type(8)))  __bf16 bf16x8;
typedef __attribute__((ext_vector_type(4)))  __bf16 bf16x4;
typedef __attribute__((ext_vector_type(8)))  float  v8f;
typedef __attribute__((ext_vector_type(2)))  int    v2i;

static constexpr int Bc = 4, Sc = 2048, Hc = 768, Ic = 3072, Lc = 4;
static constexpr int Mrows = Bc * Sc;           // 8192 token rows
static constexpr float LN_EPS = 1e-5f;

// retention scan chunking
static constexpr int CLEN = 64, CH = Sc / CLEN; // 64-elem chunks, 32 chunks

// ---- GEMM tiling: 128x128 block tile, K-step 32, 8 waves (wave32) ----------
static constexpr int BM = 128, BN = 128, BK = 32;
static constexpr int LDS_S = BK + 8;            // bf16 stride (80B, 8B-aligned)

enum { ACT_NONE = 0, ACT_SIGMOID = 1, ACT_GELU = 2 };

// ---- CDNA5 async global->LDS staging (guarded; falls back cleanly) ---------
#if defined(__has_builtin)
#  if __has_builtin(__builtin_amdgcn_global_load_async_to_lds_b64)
#    define HAS_ASYNC_LDS 1
#  endif
#endif
#ifndef HAS_ASYNC_LDS
#  define HAS_ASYNC_LDS 0
#endif

__device__ __forceinline__ void async_copy_b64(const __bf16* g, __bf16* l) {
#if HAS_ASYNC_LDS
  __builtin_amdgcn_global_load_async_to_lds_b64(
      (__attribute__((address_space(1))) v2i*)g,
      (__attribute__((address_space(3))) v2i*)l, 0, 0);
#else
  (void)g; (void)l;
#endif
}
__device__ __forceinline__ void wait_async_lds() {
#if HAS_ASYNC_LDS
#  if __has_builtin(__builtin_amdgcn_s_wait_asynccnt)
  __builtin_amdgcn_s_wait_asynccnt(0);
#  else
  asm volatile("s_wait_asynccnt 0x0" ::: "memory");
#  endif
#endif
}

// ---- vec helpers -----------------------------------------------------------
template <typename T> struct VecOf;
template <> struct VecOf<float>  { using type = float4; };
template <> struct VecOf<__bf16> { using type = bf16x4; };

__device__ __forceinline__ bf16x4 to_bf16x4(const float4& g) {
  bf16x4 p;
  p.x = (__bf16)g.x; p.y = (__bf16)g.y; p.z = (__bf16)g.z; p.w = (__bf16)g.w;
  return p;
}
__device__ __forceinline__ void store_out(float* Y, size_t i, float v) { Y[i] = v; }
__device__ __forceinline__ void store_out(__bf16* Y, size_t i, float v) { Y[i] = (__bf16)v; }

// ---------------------------------------------------------------------------
// GEMM: Y[M,N] = act(X[M,K] @ W[K,N] + bias), W pre-transposed as Wt[N,K] bf16.
// ---------------------------------------------------------------------------
template <int ACT, typename TX, typename TY>
__global__ void __launch_bounds__(256)
gemm_wmma(const TX* __restrict__ X, const __bf16* __restrict__ Wt,
          const float* __restrict__ bias, TY* __restrict__ Y,
          int K, int N)
{
  constexpr bool XBF = (sizeof(TX) == 2);       // X already bf16?

  __shared__ __bf16 lX[2][BM * LDS_S];
  __shared__ __bf16 lW[2][BN * LDS_S];

  const int tid  = threadIdx.x;
  const int lane = tid & 31;
  const int wave = tid >> 5;        // 0..7
  const int wm   = wave >> 1;       // 0..3 : 32 rows per wave
  const int wn   = wave & 1;        // 0..1 : 64 cols per wave
  const int half = lane >> 4;       // lane-half (0/1)
  const int l16  = lane & 15;

  const int m0 = blockIdx.y * BM;
  const int n0 = blockIdx.x * BN;

  // staging mapping: 4 vec4 (8-byte bf16) transfers each for X and W tiles
  const int xr[4] = { (tid + 0*256) >> 3, (tid + 1*256) >> 3,
                      (tid + 2*256) >> 3, (tid + 3*256) >> 3 };
  const int xc4   = tid & 7;
  using XV = typename VecOf<TX>::type;

  v8f zero = {};
  v8f acc[2][4];
  #pragma unroll
  for (int fm = 0; fm < 2; ++fm)
    #pragma unroll
    for (int fn = 0; fn < 4; ++fn) acc[fm][fn] = zero;

  XV xg[4];

  // ---- stage tile k0=0 into buffer 0 ----
  if constexpr (XBF && HAS_ASYNC_LDS) {
    #pragma unroll
    for (int i = 0; i < 4; ++i)
      async_copy_b64((const __bf16*)&X[(size_t)(m0 + xr[i]) * K + xc4 * 4],
                     &lX[0][xr[i] * LDS_S + xc4 * 4]);
  } else {
    #pragma unroll
    for (int i = 0; i < 4; ++i)
      xg[i] = *(const XV*)&X[(size_t)(m0 + xr[i]) * K + xc4 * 4];
  }
#if HAS_ASYNC_LDS
  #pragma unroll
  for (int i = 0; i < 4; ++i)
    async_copy_b64(&Wt[(size_t)(n0 + xr[i]) * K + xc4 * 4],
                   &lW[0][xr[i] * LDS_S + xc4 * 4]);
#else
  bf16x4 wg[4];
  #pragma unroll
  for (int i = 0; i < 4; ++i)
    wg[i] = *(const bf16x4*)&Wt[(size_t)(n0 + xr[i]) * K + xc4 * 4];
  #pragma unroll
  for (int i = 0; i < 4; ++i)
    *(bf16x4*)&lW[0][xr[i] * LDS_S + xc4 * 4] = wg[i];
#endif
  if constexpr (!(XBF && HAS_ASYNC_LDS)) {
    #pragma unroll
    for (int i = 0; i < 4; ++i) {
      if constexpr (XBF)
        *(bf16x4*)&lX[0][xr[i] * LDS_S + xc4 * 4] = xg[i];
      else
        *(bf16x4*)&lX[0][xr[i] * LDS_S + xc4 * 4] = to_bf16x4(xg[i]);
    }
  }
  wait_async_lds();
  __syncthreads();

  int buf = 0;
  for (int k0 = BK; k0 <= K; k0 += BK) {
    // ---- issue next tile's staging (targets lds[buf^1], idle since barrier)
    if (k0 < K) {
#if HAS_ASYNC_LDS
      #pragma unroll
      for (int i = 0; i < 4; ++i)
        async_copy_b64(&Wt[(size_t)(n0 + xr[i]) * K + k0 + xc4 * 4],
                       &lW[buf ^ 1][xr[i] * LDS_S + xc4 * 4]);
#endif
      if constexpr (XBF && HAS_ASYNC_LDS) {
        #pragma unroll
        for (int i = 0; i < 4; ++i)
          async_copy_b64((const __bf16*)&X[(size_t)(m0 + xr[i]) * K + k0 + xc4 * 4],
                         &lX[buf ^ 1][xr[i] * LDS_S + xc4 * 4]);
      } else {
        #pragma unroll
        for (int i = 0; i < 4; ++i)
          xg[i] = *(const XV*)&X[(size_t)(m0 + xr[i]) * K + k0 + xc4 * 4];
      }
#if !HAS_ASYNC_LDS
      #pragma unroll
      for (int i = 0; i < 4; ++i)
        wg[i] = *(const bf16x4*)&Wt[(size_t)(n0 + xr[i]) * K + k0 + xc4 * 4];
#endif
    }

    // ---- compute on lds[buf] while staging is in flight ----
    // A frags: ISA 16-bit 16x32 A layout -> two contiguous 8-elem groups
    v16bf afr[2];
    #pragma unroll
    for (int fm = 0; fm < 2; ++fm) {
      const __bf16* ap = &lX[buf][(wm * 32 + fm * 16 + l16) * LDS_S];
      bf16x8 alo = *(const bf16x8*)&ap[8 * half];
      bf16x8 ahi = *(const bf16x8*)&ap[16 + 8 * half];
      afr[fm] = __builtin_shufflevector(alo, ahi, 0,1,2,3,4,5,6,7,
                                        8,9,10,11,12,13,14,15);
    }
    // B frags: same layout as A -> two 8-elem groups at k = 8*half and 16 + 8*half
    v16bf bfr[4];
    #pragma unroll
    for (int fn = 0; fn < 4; ++fn) {
      const __bf16* bp = &lW[buf][(wn * 64 + fn * 16 + l16) * LDS_S + 8 * half];
      bf16x8 blo = *(const bf16x8*)&bp[0];
      bf16x8 bhi = *(const bf16x8*)&bp[16];
      bfr[fn] = __builtin_shufflevector(blo, bhi, 0,1,2,3,4,5,6,7,
                                        8,9,10,11,12,13,14,15);
    }
    #pragma unroll
    for (int fm = 0; fm < 2; ++fm)
      #pragma unroll
      for (int fn = 0; fn < 4; ++fn)
        acc[fm][fn] = __builtin_amdgcn_wmma_f32_16x16x32_bf16(
            false, afr[fm], false, bfr[fn], (short)0, acc[fm][fn], false, false);

    // ---- finish staging the next buffer ----
    if (k0 < K) {
      if constexpr (!(XBF && HAS_ASYNC_LDS)) {
        #pragma unroll
        for (int i = 0; i < 4; ++i) {
          if constexpr (XBF)
            *(bf16x4*)&lX[buf ^ 1][xr[i] * LDS_S + xc4 * 4] = xg[i];
          else
            *(bf16x4*)&lX[buf ^ 1][xr[i] * LDS_S + xc4 * 4] = to_bf16x4(xg[i]);
        }
      }
#if !HAS_ASYNC_LDS
      #pragma unroll
      for (int i = 0; i < 4; ++i)
        *(bf16x4*)&lW[buf ^ 1][xr[i] * LDS_S + xc4 * 4] = wg[i];
#endif
      wait_async_lds();
      __syncthreads();
      buf ^= 1;
    }
  }

  // ---- epilogue: bias + activation + store (C layout: row = i + 8*half) ----
  #pragma unroll
  for (int fm = 0; fm < 2; ++fm) {
    #pragma unroll
    for (int fn = 0; fn < 4; ++fn) {
      int col = n0 + wn * 64 + fn * 16 + l16;
      float bv = bias[col];
      #pragma unroll
      for (int i = 0; i < 8; ++i) {
        int row = m0 + wm * 32 + fm * 16 + half * 8 + i;
        float v = acc[fm][fn][i] + bv;
        if constexpr (ACT == ACT_SIGMOID) {
          v = 1.0f / (1.0f + __expf(-v));
        } else if constexpr (ACT == ACT_GELU) {
          float t = tanhf(0.7978845608028654f * (v + 0.044715f * v * v * v));
          v = 0.5f * v * (1.0f + t);
        }
        store_out(Y, (size_t)row * N + col, v);
      }
    }
  }
}

// ---------------------------------------------------------------------------
// One-time per call: W[K,N] fp32 -> Wt[N,K] bf16 (LDS-tiled 32x32 transpose)
// ---------------------------------------------------------------------------
__global__ void __launch_bounds__(256)
transpose_to_bf16(const float* __restrict__ W, __bf16* __restrict__ Wt,
                  int K, int N)
{
  __shared__ float t[32][33];
  int tx = threadIdx.x & 31, ty = threadIdx.x >> 5;      // ty: 0..7
  int n0 = blockIdx.x * 32, k0 = blockIdx.y * 32;
  #pragma unroll
  for (int i = 0; i < 4; ++i)
    t[ty + 8 * i][tx] = W[(size_t)(k0 + ty + 8 * i) * N + (n0 + tx)];
  __syncthreads();
  #pragma unroll
  for (int i = 0; i < 4; ++i)
    Wt[(size_t)(n0 + ty + 8 * i) * K + (k0 + tx)] = (__bf16)t[tx][ty + 8 * i];
}

// ---------------------------------------------------------------------------
// Embedding gather + LayerNorm. One 256-thread block per token row (H=768).
// ---------------------------------------------------------------------------
__global__ void __launch_bounds__(256)
embed_ln(const int* __restrict__ ids, const float* __restrict__ wemb,
         const float* __restrict__ pemb, const float* __restrict__ sc,
         const float* __restrict__ bs, float* __restrict__ out)
{
  int row  = blockIdx.x;
  int sidx = row % Sc;
  int id   = ids[row];
  const float* wp = wemb + (size_t)id * Hc;
  const float* pp = pemb + (size_t)sidx * Hc;

  __shared__ float rs[256], rq[256];
  int t = threadIdx.x;
  float vals[3], sum = 0.f, sq = 0.f;
  #pragma unroll
  for (int i = 0; i < 3; ++i) {
    int j = t + i * 256;
    float v = wp[j] + pp[j];
    vals[i] = v; sum += v; sq += v * v;
  }
  rs[t] = sum; rq[t] = sq;
  __syncthreads();
  for (int off = 128; off > 0; off >>= 1) {
    if (t < off) { rs[t] += rs[t + off]; rq[t] += rq[t + off]; }
    __syncthreads();
  }
  float mu  = rs[0] * (1.0f / Hc);
  float var = rq[0] * (1.0f / Hc) - mu * mu;
  float inv = rsqrtf(var + LN_EPS);
  float* op = out + (size_t)row * Hc;
  #pragma unroll
  for (int i = 0; i < 3; ++i) {
    int j = t + i * 256;
    op[j] = (vals[i] - mu) * inv * sc[j] + bs[j];
  }
}

// ---------------------------------------------------------------------------
// out = LayerNorm(a + b) (b may be null for the final LN).
// ---------------------------------------------------------------------------
__global__ void __launch_bounds__(256)
add_ln(const float* __restrict__ a, const float* __restrict__ b,
       const float* __restrict__ sc, const float* __restrict__ bs,
       float* __restrict__ out)
{
  int row = blockIdx.x;
  const float* ap = a + (size_t)row * Hc;
  const float* bp = b ? (b + (size_t)row * Hc) : nullptr;

  __shared__ float rs[256], rq[256];
  int t = threadIdx.x;
  float vals[3], sum = 0.f, sq = 0.f;
  #pragma unroll
  for (int i = 0; i < 3; ++i) {
    int j = t + i * 256;
    float v = ap[j] + (bp ? bp[j] : 0.0f);
    vals[i] = v; sum += v; sq += v * v;
  }
  rs[t] = sum; rq[t] = sq;
  __syncthreads();
  for (int off = 128; off > 0; off >>= 1) {
    if (t < off) { rs[t] += rs[t + off]; rq[t] += rq[t + off]; }
    __syncthreads();
  }
  float mu  = rs[0] * (1.0f / Hc);
  float var = rq[0] * (1.0f / Hc) - mu * mu;
  float inv = rsqrtf(var + LN_EPS);
  float* op = out + (size_t)row * Hc;
  #pragma unroll
  for (int i = 0; i < 3; ++i) {
    int j = t + i * 256;
    op[j] = (vals[i] - mu) * inv * sc[j] + bs[j];
  }
}

// ---------------------------------------------------------------------------
// Chunked parallel retention scan: r_t = 0.5*(s_t + r_{t-1}).
// Pass1: per-(b,h,chunk) local scan (r_in=0), carry -> cbuf.
// Pass2: per-(b,h) serial carry propagation across 32 chunks (r_in per chunk).
// Pass3: add correction 0.5^(t+1) * r_in to each chunk's elements.
// ---------------------------------------------------------------------------
__global__ void __launch_bounds__(256)
scan_pass1(float* __restrict__ s, float* __restrict__ cbuf)
{
  int idx = blockIdx.x * blockDim.x + threadIdx.x;   // [0, B*CH*H)
  int hh = idx % Hc;
  int c  = (idx / Hc) % CH;
  int b  = idx / (Hc * CH);
  float* p = s + (size_t)b * Sc * Hc + (size_t)c * CLEN * Hc + hh;
  float r = 0.0f;
  for (int t = 0; t < CLEN; ++t) {
    r = 0.5f * (p[(size_t)t * Hc] + r);
    p[(size_t)t * Hc] = r;
  }
  cbuf[((size_t)b * CH + c) * Hc + hh] = r;          // local carry
}

__global__ void __launch_bounds__(256)
scan_pass2(float* __restrict__ cbuf)
{
  int idx = blockIdx.x * blockDim.x + threadIdx.x;   // [0, B*H)
  int hh = idx % Hc, b = idx / Hc;
  float rin = 0.0f;
  for (int c = 0; c < CH; ++c) {
    size_t o = ((size_t)b * CH + c) * Hc + hh;
    float lc = cbuf[o];
    cbuf[o] = rin;                                   // r_in for chunk c
    rin = lc + 0x1p-64f * rin;                       // 0.5^CLEN carry decay
  }
}

__global__ void __launch_bounds__(256)
scan_pass3(float* __restrict__ s, const float* __restrict__ cbuf)
{
  int idx = blockIdx.x * blockDim.x + threadIdx.x;   // [0, B*CH*H)
  int hh = idx % Hc;
  int c  = (idx / Hc) % CH;
  int b  = idx / (Hc * CH);
  float rin = cbuf[((size_t)b * CH + c) * Hc + hh];
  if (rin == 0.0f) return;                           // chunk 0 / zero carry
  float* p = s + (size_t)b * Sc * Hc + (size_t)c * CLEN * Hc + hh;
  float f = 0.5f;
  // 0.5^25 ~ 3e-8: beyond 32 steps the correction is below fp32 ulp
  for (int t = 0; t < 32; ++t) {
    p[(size_t)t * Hc] += f * rin;
    f *= 0.5f;
  }
}

// ---------------------------------------------------------------------------
extern "C" void kernel_launch(void* const* d_in, const int* in_sizes, int n_in,
                              void* d_out, int out_size, void* d_ws, size_t ws_size,
                              hipStream_t stream)
{
  (void)in_sizes; (void)n_in; (void)out_size; (void)ws_size;

  const int*   ids  = (const int*)  d_in[0];
  const float* wemb = (const float*)d_in[1];
  const float* pemb = (const float*)d_in[2];
  const float* es   = (const float*)d_in[3];
  const float* ebv  = (const float*)d_in[4];
  const float* retW = (const float*)d_in[5];   // [L,H,H]
  const float* retb = (const float*)d_in[6];   // [L,H]
  const float* l1s  = (const float*)d_in[7];
  const float* l1b  = (const float*)d_in[8];
  const float* W1   = (const float*)d_in[9];   // [L,H,I]
  const float* b1   = (const float*)d_in[10];  // [L,I]
  const float* W2   = (const float*)d_in[11];  // [L,I,H]
  const float* b2   = (const float*)d_in[12];  // [L,H]
  const float* l2s  = (const float*)d_in[13];
  const float* l2b  = (const float*)d_in[14];
  const float* fs   = (const float*)d_in[15];
  const float* fb   = (const float*)d_in[16];

  // workspace layout (~169 MB):
  //   x | h | s : [M,H] fp32 ; f1 : [M,I] bf16 ;
  //   WtR/Wt1/Wt2 : bf16 transposed weights ; cbuf : [B,CH,H] fp32
  float*  x   = (float*)d_ws;
  float*  h   = x  + (size_t)Mrows * Hc;
  float*  sb  = h  + (size_t)Mrows * Hc;
  __bf16* f1  = (__bf16*)(sb + (size_t)Mrows * Hc);
  __bf16* WtR = f1  + (size_t)Mrows * Ic;
  __bf16* Wt1 = WtR + (size_t)Lc * Hc * Hc;
  __bf16* Wt2 = Wt1 + (size_t)Lc * Ic * Hc;
  float*  cb  = (float*)(Wt2 + (size_t)Lc * Hc * Ic);

  dim3 blk(256);

  // ---- weight transpose + bf16 convert ----
  for (int l = 0; l < Lc; ++l) {
    transpose_to_bf16<<<dim3(Hc / 32, Hc / 32), blk, 0, stream>>>(
        retW + (size_t)l * Hc * Hc, WtR + (size_t)l * Hc * Hc, Hc, Hc);
    transpose_to_bf16<<<dim3(Ic / 32, Hc / 32), blk, 0, stream>>>(
        W1 + (size_t)l * Hc * Ic, Wt1 + (size_t)l * Ic * Hc, Hc, Ic);
    transpose_to_bf16<<<dim3(Hc / 32, Ic / 32), blk, 0, stream>>>(
        W2 + (size_t)l * Ic * Hc, Wt2 + (size_t)l * Hc * Ic, Ic, Hc);
  }

  embed_ln<<<Mrows, blk, 0, stream>>>(ids, wemb, pemb, es, ebv, x);

  const int scan_blocks  = (Bc * CH * Hc) / 256;   // 384
  const int carry_blocks = (Bc * Hc) / 256;        // 12

  for (int l = 0; l < Lc; ++l) {
    // s = sigmoid(x @ retW + retb)
    gemm_wmma<ACT_SIGMOID, float, float>
        <<<dim3(Hc / BN, Mrows / BM), blk, 0, stream>>>(
        x, WtR + (size_t)l * Hc * Hc, retb + (size_t)l * Hc, sb, Hc, Hc);
    // retention scan (in place, 3-pass parallel)
    scan_pass1<<<scan_blocks, blk, 0, stream>>>(sb, cb);
    scan_pass2<<<carry_blocks, blk, 0, stream>>>(cb);
    scan_pass3<<<scan_blocks, blk, 0, stream>>>(sb, cb);
    // h = LN(x + retention)
    add_ln<<<Mrows, blk, 0, stream>>>(x, sb, l1s + (size_t)l * Hc,
                                      l1b + (size_t)l * Hc, h);
    // f1 = gelu(h @ W1 + b1)   (bf16 output)
    gemm_wmma<ACT_GELU, float, __bf16>
        <<<dim3(Ic / BN, Mrows / BM), blk, 0, stream>>>(
        h, Wt1 + (size_t)l * Ic * Hc, b1 + (size_t)l * Ic, f1, Hc, Ic);
    // s = f1 @ W2 + b2         (bf16 input)
    gemm_wmma<ACT_NONE, __bf16, float>
        <<<dim3(Hc / BN, Mrows / BM), blk, 0, stream>>>(
        f1, Wt2 + (size_t)l * Hc * Ic, b2 + (size_t)l * Hc, sb, Ic, Hc);
    // x = LN(h + ffn)
    add_ln<<<Mrows, blk, 0, stream>>>(h, sb, l2s + (size_t)l * Hc,
                                      l2b + (size_t)l * Hc, x);
  }

  add_ln<<<Mrows, blk, 0, stream>>>(x, nullptr, fs, fb, (float*)d_out);
}
